// LinearizedContext_721554506168
// MI455X (gfx1250) — matrix operand order repaired and checked by reading, not compiled, not run on hardware
//
#include <hip/hip_runtime.h>
#include <hip/hip_bf16.h>
#include <stdint.h>

typedef __attribute__((ext_vector_type(16))) _Float16 v16h;
typedef __attribute__((ext_vector_type(8)))  _Float16 v8h;
typedef __attribute__((ext_vector_type(8)))  float    v8f;
typedef __attribute__((ext_vector_type(4)))  int      v4i;

#define DEV __device__ __forceinline__

// ---- optional CDNA5 async global->LDS path (guarded; falls back to b128) ----
#if defined(__has_builtin)
#  if __has_builtin(__builtin_amdgcn_global_load_async_to_lds_b128)
#    define USE_ASYNC_LDS 1
#  endif
#  if __has_builtin(__builtin_amdgcn_s_wait_asynccnt)
#    define HAVE_WAIT_ASYNC_BUILTIN 1
#  endif
#endif
#ifndef USE_ASYNC_LDS
#  define USE_ASYNC_LDS 0
#endif
#ifndef HAVE_WAIT_ASYNC_BUILTIN
#  define HAVE_WAIT_ASYNC_BUILTIN 0
#endif

// copy 16B global -> LDS (async-to-lds when available)
DEV void copy16_g2l(_Float16* l, const _Float16* g) {
#if USE_ASYNC_LDS
  // diag-probed signature: param 1 is v4i* (flat); order = (global src, lds dst)
  __builtin_amdgcn_global_load_async_to_lds_b128(
      (v4i*)const_cast<_Float16*>(g), (v4i*)l, 0, 0);
#else
  *(v8h*)l = *(const v8h*)g;
#endif
}

DEV void wait_async0() {
#if USE_ASYNC_LDS
#  if HAVE_WAIT_ASYNC_BUILTIN
  __builtin_amdgcn_s_wait_asynccnt(0);
#  else
  asm volatile("s_wait_asynccnt 0" ::: "memory");
#  endif
#endif
}

union F16Frag { v16h v; v8h h8[2]; _Float16 h[16]; };

// A-fragment (16x32, MxK): lane L -> row = L%16; halves 0-7 hold K = k0+8h..,
// halves 8-15 hold K = k0+16+8h.. (h = L/16).  Two 16B loads.
DEV v16h load_afrag(const _Float16* base, int pitch, int row, int k0, int lane) {
  const int hh = (lane >> 4) & 1;
  const _Float16* p = base + (size_t)row * pitch + k0 + 8 * hh;
  F16Frag f;
  f.h8[0] = *(const v8h*)(p);
  f.h8[1] = *(const v8h*)(p + 16);
  return f.v;
}

// B-fragment (32x16, KxN) read from an n-major source (row = n, k contiguous):
// lane L -> col n = L%16; halves 0-7 hold K = k0+8h.., halves 8-15 hold K = k0+16+8h..
DEV v16h load_bfrag(const _Float16* base, int pitch, int col, int k0, int lane) {
  const int hh = (lane >> 4) & 1;
  const _Float16* p = base + (size_t)col * pitch + k0 + 8 * hh;
  F16Frag f;
  f.h8[0] = *(const v8h*)(p);
  f.h8[1] = *(const v8h*)(p + 16);
  return f.v;
}

DEV v8f wmma_f16(v16h a, v16h b, v8f c) {
  return __builtin_amdgcn_wmma_f32_16x16x32_f16(false, a, false, b, (short)0, c,
                                                false, false);
}

// ---------------------------------------------------------------------------
// prep: X1[8192 x 512] f16 = concat(emb2[preds], obj_ctx), zero-padded cols.
// ---------------------------------------------------------------------------
__global__ void prep_x1(const float* __restrict__ obj_ctx, const float* __restrict__ emb2,
                        const int* __restrict__ preds, _Float16* __restrict__ X1) {
  int idx = blockIdx.x * 256 + threadIdx.x;   // over 8192*512
  int row = idx >> 9, col = idx & 511;
  float v = 0.f;
  if (col < 200)      v = emb2[(size_t)preds[row] * 200 + col];
  else if (col < 456) v = obj_ctx[(size_t)row * 256 + (col - 200)];
  X1[idx] = (_Float16)v;
}

// W f32 [Nr x Kr] -> f16 [Npad x Kpad] n-major, zero padded.
__global__ void convert_w(const float* __restrict__ W, _Float16* __restrict__ out,
                          int Nr, int Kr, int Kpad) {
  int idx = blockIdx.x * 256 + threadIdx.x;   // over Npad*Kpad
  int n = idx / Kpad, k = idx % Kpad;
  out[idx] = (n < Nr && k < Kr) ? (_Float16)W[(size_t)n * Kr + k] : (_Float16)0.f;
}

// ---------------------------------------------------------------------------
// WMMA GEMM: C[M x Npad] = A[M x Kpad](f16 row-major) * B^T, B n-major
// f16 [Npad x Kpad].  Epilogue (compile-time flags): +bias, +residual(f16),
// f32 store (ldo), f16 store (ldh), transposed f16 store [Npad x M] (ldt).
// Block 256 = 8 wave32; tile 128(M) x 128(N); each wave 32x64 = 8 WMMA/k-step.
// ---------------------------------------------------------------------------
template<bool BIAS, bool RES, bool OF, bool OH, bool OT>
__global__ void gemm_wmma(const _Float16* __restrict__ A, int lda,
                          const _Float16* __restrict__ B, int ldb,
                          const float* __restrict__ bias, int nbias,
                          const _Float16* __restrict__ resid, int ldr,
                          float* __restrict__ outF, int ldo,
                          _Float16* __restrict__ outH, int ldh,
                          _Float16* __restrict__ outT, int ldt,
                          int Kpad) {
  __shared__ alignas(16) _Float16 As[128 * 32];
  __shared__ alignas(16) _Float16 Bs[128 * 32];
  const int tid = threadIdx.x;
  const int lane = tid & 31, wave = tid >> 5;
  const int hh = (lane >> 4) & 1, nloc = lane & 15;
  const int mblk = blockIdx.y * 128, nblk = blockIdx.x * 128;
  const int wm = (wave >> 1) * 32, wn = (wave & 1) * 64;
  v8f acc[2][4] = {};

  for (int k0 = 0; k0 < Kpad; k0 += 32) {
    __syncthreads();
#pragma unroll
    for (int t = 0; t < 2; ++t) {             // 512 16B-chunks per tile
      int w = tid + t * 256;
      int r = w >> 2, c = (w & 3) * 8;
      copy16_g2l(As + r * 32 + c, A + (size_t)(mblk + r) * lda + k0 + c);
      copy16_g2l(Bs + r * 32 + c, B + (size_t)(nblk + r) * ldb + k0 + c);
    }
    if (k0 + 32 < Kpad) {                     // prefetch next K slice
      __builtin_prefetch(A + (size_t)(mblk + (tid >> 1)) * lda + k0 + 32 + (tid & 1) * 16, 0, 1);
      __builtin_prefetch(B + (size_t)(nblk + (tid >> 1)) * ldb + k0 + 32 + (tid & 1) * 16, 0, 1);
    }
    wait_async0();
    __syncthreads();
    v16h a0 = load_afrag(As, 32, wm + nloc, 0, lane);
    v16h a1 = load_afrag(As, 32, wm + 16 + nloc, 0, lane);
#pragma unroll
    for (int j = 0; j < 4; ++j) {
      v16h b = load_bfrag(Bs, 32, wn + j * 16 + nloc, 0, lane);
      acc[0][j] = wmma_f16(a0, b, acc[0][j]);
      acc[1][j] = wmma_f16(a1, b, acc[1][j]);
    }
  }

#pragma unroll
  for (int i = 0; i < 2; ++i)
#pragma unroll
    for (int j = 0; j < 4; ++j) {
      const int col = nblk + wn + j * 16 + nloc;
      const int rbase = mblk + wm + i * 16 + 8 * hh;      // C layout: row = v+8h
      float bv = 0.f;
      if constexpr (BIAS) bv = (col < nbias) ? bias[col] : 0.f;
      float vals[8];
#pragma unroll
      for (int v = 0; v < 8; ++v) {
        float x = acc[i][j][v] + bv;
        if constexpr (RES) x += (float)resid[(size_t)(rbase + v) * ldr + col];
        vals[v] = x;
      }
      if constexpr (OF) {
#pragma unroll
        for (int v = 0; v < 8; ++v)
          outF[(size_t)(rbase + v) * ldo + col] = vals[v];
      }
      if constexpr (OH) {
#pragma unroll
        for (int v = 0; v < 8; ++v)
          outH[(size_t)(rbase + v) * ldh + col] = (_Float16)vals[v];
      }
      if constexpr (OT) {                     // contiguous v8h per lane
        v8h pk;
#pragma unroll
        for (int v = 0; v < 8; ++v) pk[v] = (_Float16)vals[v];
        *(v8h*)(outT + (size_t)col * ldt + rbase) = pk;
      }
    }
}

// ---------------------------------------------------------------------------
// Attention pass 1: row max & exp-sum of Q K^T.  Block = 128 rows (16/wave).
// ---------------------------------------------------------------------------
template<int Dp>
__global__ void attn_pass1(const _Float16* __restrict__ Q, const _Float16* __restrict__ K,
                           float* __restrict__ Mrow, float* __restrict__ Srow, int N) {
  __shared__ alignas(16) _Float16 Ks[16 * Dp];
  const int tid = threadIdx.x;
  const int lane = tid & 31, wave = tid >> 5;
  const int r0 = blockIdx.x * 128 + wave * 16;
  const int hh = (lane >> 4) & 1;
  constexpr int NK = Dp / 32;
  constexpr int KCH = 2 * Dp;                 // 16B chunks in 16 x Dp tile

  v16h qf[NK];
#pragma unroll
  for (int kk = 0; kk < NK; ++kk)
    qf[kk] = load_afrag(Q, Dp, r0 + (lane & 15), kk * 32, lane);

  float mrun[8], srun[8];
#pragma unroll
  for (int v = 0; v < 8; ++v) { mrun[v] = -1e30f; srun[v] = 0.f; }

  for (int j0 = 0; j0 < N; j0 += 16) {
    __syncthreads();
    for (int w = tid; w < KCH; w += 256)      // contiguous 16 rows of K
      copy16_g2l(Ks + w * 8, K + (size_t)j0 * Dp + w * 8);
    wait_async0();
    __syncthreads();
    v8f s = {};
#pragma unroll
    for (int kk = 0; kk < NK; ++kk)
      s = wmma_f16(qf[kk], load_bfrag(Ks, Dp, lane & 15, kk * 32, lane), s);
#pragma unroll
    for (int v = 0; v < 8; ++v) {
      float x = s[v];
      float mx = x;                           // reduce over 16 cols (lanes)
      mx = fmaxf(mx, __shfl_xor(mx, 1, 32));
      mx = fmaxf(mx, __shfl_xor(mx, 2, 32));
      mx = fmaxf(mx, __shfl_xor(mx, 4, 32));
      mx = fmaxf(mx, __shfl_xor(mx, 8, 32));
      float mnew = fmaxf(mrun[v], mx);
      float e = __expf(x - mnew);
      e += __shfl_xor(e, 1, 32);
      e += __shfl_xor(e, 2, 32);
      e += __shfl_xor(e, 4, 32);
      e += __shfl_xor(e, 8, 32);
      srun[v] = srun[v] * __expf(mrun[v] - mnew) + e;
      mrun[v] = mnew;
    }
  }
  if ((lane & 15) == 0) {
#pragma unroll
    for (int v = 0; v < 8; ++v) {
      Mrow[r0 + v + 8 * hh] = mrun[v];
      Srow[r0 + v + 8 * hh] = srun[v];
    }
  }
}

// ---------------------------------------------------------------------------
// Attention pass 2: O = (exp(QK^T - m)/s) V.  V supplied transposed [Dp x N].
// ---------------------------------------------------------------------------
template<int Dp>
__global__ void attn_pass2(const _Float16* __restrict__ Q, const _Float16* __restrict__ K,
                           const _Float16* __restrict__ Vt,
                           const float* __restrict__ Mrow, const float* __restrict__ Srow,
                           _Float16* __restrict__ O, int N) {
  __shared__ alignas(16) _Float16 Ks[32 * Dp];
  __shared__ alignas(16) _Float16 Vs[Dp * 32];
  __shared__ alignas(16) _Float16 Ps[8 * 16 * 32];
  const int tid = threadIdx.x;
  const int lane = tid & 31, wave = tid >> 5;
  const int r0 = blockIdx.x * 128 + wave * 16;
  const int hh = (lane >> 4) & 1;
  constexpr int NK = Dp / 32, ND = Dp / 16;

  v16h qf[NK];
#pragma unroll
  for (int kk = 0; kk < NK; ++kk)
    qf[kk] = load_afrag(Q, Dp, r0 + (lane & 15), kk * 32, lane);

  float mrow[8], sinv[8];
#pragma unroll
  for (int v = 0; v < 8; ++v) {
    mrow[v] = Mrow[r0 + v + 8 * hh];
    sinv[v] = 1.f / Srow[r0 + v + 8 * hh];
  }

  v8f acc[ND] = {};
  _Float16* Pw = Ps + wave * 512;             // per-wave 16x32 staging

  for (int j0 = 0; j0 < N; j0 += 32) {
    __syncthreads();
    for (int w = tid; w < 4 * Dp; w += 256)   // K tile: 32 contiguous rows
      copy16_g2l(Ks + w * 8, K + (size_t)j0 * Dp + w * 8);
    for (int w = tid; w < 4 * Dp; w += 256) { // Vt tile: Dp rows x 32, pitch N
      int r = w >> 2, c = (w & 3) * 8;
      copy16_g2l(Vs + r * 32 + c, Vt + (size_t)r * N + j0 + c);
    }
    wait_async0();
    __syncthreads();

#pragma unroll
    for (int t = 0; t < 2; ++t) {
      v8f s = {};
#pragma unroll
      for (int kk = 0; kk < NK; ++kk)
        s = wmma_f16(qf[kk],
                     load_bfrag(Ks, Dp, t * 16 + (lane & 15), kk * 32, lane), s);
#pragma unroll
      for (int v = 0; v < 8; ++v)
        Pw[(v + 8 * hh) * 32 + t * 16 + (lane & 15)] =
            (_Float16)__expf(s[v] - mrow[v]);
    }
    // same-wave LDS write -> cross-lane read (DS in-order; make it explicit)
    asm volatile("s_wait_dscnt 0" ::: "memory");
    v16h pf = load_afrag(Pw, 32, lane & 15, 0, lane);
#pragma unroll
    for (int t2 = 0; t2 < ND; ++t2)
      acc[t2] = wmma_f16(pf, load_bfrag(Vs, 32, t2 * 16 + (lane & 15), 0, lane),
                         acc[t2]);
  }

#pragma unroll
  for (int t2 = 0; t2 < ND; ++t2)
#pragma unroll
    for (int v = 0; v < 8; ++v)
      O[(size_t)(r0 + v + 8 * hh) * Dp + t2 * 16 + (lane & 15)] =
          (_Float16)(acc[t2][v] * sinv[v]);
}

// ---------------------------------------------------------------------------
extern "C" void kernel_launch(void* const* d_in, const int* in_sizes, int n_in,
                              void* d_out, int out_size, void* d_ws, size_t ws_size,
                              hipStream_t stream) {
  (void)in_sizes; (void)n_in; (void)out_size; (void)ws_size;
  const int N = 8192;
  const float* obj_ctx = (const float*)d_in[0];
  const float* emb2    = (const float*)d_in[1];
  const float* r1_Ww = (const float*)d_in[2];  const float* r1_Wb = (const float*)d_in[3];
  const float* r1_Uw = (const float*)d_in[4];  const float* r1_Ub = (const float*)d_in[5];
  const float* r1_Hw = (const float*)d_in[6];  const float* r1_Hb = (const float*)d_in[7];
  const float* r1_f0w = (const float*)d_in[8]; const float* r1_f0b = (const float*)d_in[9];
  const float* r1_f1w = (const float*)d_in[10];const float* r1_f1b = (const float*)d_in[11];
  const float* r2_Ww = (const float*)d_in[12]; const float* r2_Wb = (const float*)d_in[13];
  const float* r2_Uw = (const float*)d_in[14]; const float* r2_Ub = (const float*)d_in[15];
  const float* r2_Hw = (const float*)d_in[16]; const float* r2_Hb = (const float*)d_in[17];
  const float* r2_f0w = (const float*)d_in[18];const float* r2_f0b = (const float*)d_in[19];
  const float* r2_f1w = (const float*)d_in[20];const float* r2_f1b = (const float*)d_in[21];
  const int* obj_preds = (const int*)d_in[22];

  char* wp = (char*)d_ws;
  auto alloc = [&](size_t nbytes) {
    char* p = wp; wp += (nbytes + 255) & ~(size_t)255; return p;
  };
  _Float16* X1   = (_Float16*)alloc((size_t)N * 512 * 2);   // padded concat input
  _Float16* W1q  = (_Float16*)alloc((size_t)256 * 512 * 2);
  _Float16* W1u  = (_Float16*)alloc((size_t)256 * 512 * 2);
  _Float16* W1h  = (_Float16*)alloc((size_t)256 * 512 * 2);
  _Float16* W1f0 = (_Float16*)alloc((size_t)512 * 256 * 2);
  _Float16* W1f1 = (_Float16*)alloc((size_t)256 * 512 * 2);
  _Float16* W2q  = (_Float16*)alloc((size_t)128 * 256 * 2);
  _Float16* W2u  = (_Float16*)alloc((size_t)128 * 256 * 2);
  _Float16* W2h  = (_Float16*)alloc((size_t)128 * 256 * 2);
  _Float16* W2f0 = (_Float16*)alloc((size_t)256 * 128 * 2);
  _Float16* W2f1 = (_Float16*)alloc((size_t)8192 * 256 * 2);
  _Float16* q1   = (_Float16*)alloc((size_t)N * 256 * 2);
  _Float16* k1   = (_Float16*)alloc((size_t)N * 256 * 2);
  _Float16* v1T  = (_Float16*)alloc((size_t)256 * N * 2);
  float*    m1   = (float*)alloc((size_t)N * 4);
  float*    s1   = (float*)alloc((size_t)N * 4);
  _Float16* o1   = (_Float16*)alloc((size_t)N * 256 * 2);
  _Float16* O1   = (_Float16*)alloc((size_t)N * 512 * 2);
  _Float16* X2   = (_Float16*)alloc((size_t)N * 256 * 2);
  _Float16* q2   = (_Float16*)alloc((size_t)N * 128 * 2);
  _Float16* k2   = (_Float16*)alloc((size_t)N * 128 * 2);
  _Float16* v2T  = (_Float16*)alloc((size_t)128 * N * 2);
  float*    m2   = (float*)alloc((size_t)N * 4);
  float*    s2   = (float*)alloc((size_t)N * 4);
  _Float16* o2   = (_Float16*)alloc((size_t)N * 128 * 2);
  _Float16* ec16 = (_Float16*)alloc((size_t)N * 256 * 2);
  float* edge_ctx = (float*)d_out;
  float* edge_rep = (float*)d_out + (size_t)N * 256;

  // 1. build X1 and convert all weights to padded f16 (n-major)
  prep_x1<<<N * 512 / 256, 256, 0, stream>>>(obj_ctx, emb2, obj_preds, X1);
  convert_w<<<256 * 512 / 256, 256, 0, stream>>>(r1_Ww, W1q, 228, 456, 512);
  convert_w<<<256 * 512 / 256, 256, 0, stream>>>(r1_Uw, W1u, 228, 456, 512);
  convert_w<<<256 * 512 / 256, 256, 0, stream>>>(r1_Hw, W1h, 228, 456, 512);
  convert_w<<<512 * 256 / 256, 256, 0, stream>>>(r1_f0w, W1f0, 456, 228, 256);
  convert_w<<<256 * 512 / 256, 256, 0, stream>>>(r1_f1w, W1f1, 256, 456, 512);
  convert_w<<<128 * 256 / 256, 256, 0, stream>>>(r2_Ww, W2q, 128, 256, 256);
  convert_w<<<128 * 256 / 256, 256, 0, stream>>>(r2_Uw, W2u, 128, 256, 256);
  convert_w<<<128 * 256 / 256, 256, 0, stream>>>(r2_Hw, W2h, 128, 256, 256);
  convert_w<<<256 * 128 / 256, 256, 0, stream>>>(r2_f0w, W2f0, 256, 128, 128);
  convert_w<<<8192 * 256 / 256, 256, 0, stream>>>(r2_f1w, W2f1, 8192, 256, 256);

  const dim3 blk(256);
  // 2. RE1 q/k/v  (M=8192, Npad=256, Kpad=512)
  gemm_wmma<true,false,false,true,false><<<dim3(2, 64), blk, 0, stream>>>(
      X1, 512, W1q, 512, r1_Wb, 228, nullptr, 0, nullptr, 0, q1, 256, nullptr, 0, 512);
  gemm_wmma<true,false,false,true,false><<<dim3(2, 64), blk, 0, stream>>>(
      X1, 512, W1u, 512, r1_Ub, 228, nullptr, 0, nullptr, 0, k1, 256, nullptr, 0, 512);
  gemm_wmma<true,false,false,false,true><<<dim3(2, 64), blk, 0, stream>>>(
      X1, 512, W1h, 512, r1_Hb, 228, nullptr, 0, nullptr, 0, nullptr, 0, v1T, N, 512);
  // 3. RE1 attention (Dp = 256)
  attn_pass1<256><<<64, blk, 0, stream>>>(q1, k1, m1, s1, N);
  attn_pass2<256><<<64, blk, 0, stream>>>(q1, k1, v1T, m1, s1, o1, N);
  // 4. RE1 fc0 (residual) and fc1
  gemm_wmma<true,true,false,true,false><<<dim3(4, 64), blk, 0, stream>>>(
      o1, 256, W1f0, 256, r1_f0b, 456, X1, 512, nullptr, 0, O1, 512, nullptr, 0, 256);
  gemm_wmma<true,false,false,true,false><<<dim3(2, 64), blk, 0, stream>>>(
      O1, 512, W1f1, 512, r1_f1b, 256, nullptr, 0, nullptr, 0, X2, 256, nullptr, 0, 512);
  // 5. RE2 q/k/v (M=8192, Npad=128, Kpad=256)
  gemm_wmma<true,false,false,true,false><<<dim3(1, 64), blk, 0, stream>>>(
      X2, 256, W2q, 256, r2_Wb, 128, nullptr, 0, nullptr, 0, q2, 128, nullptr, 0, 256);
  gemm_wmma<true,false,false,true,false><<<dim3(1, 64), blk, 0, stream>>>(
      X2, 256, W2u, 256, r2_Ub, 128, nullptr, 0, nullptr, 0, k2, 128, nullptr, 0, 256);
  gemm_wmma<true,false,false,false,true><<<dim3(1, 64), blk, 0, stream>>>(
      X2, 256, W2h, 256, r2_Hb, 128, nullptr, 0, nullptr, 0, nullptr, 0, v2T, N, 256);
  // 6. RE2 attention (Dp = 128)
  attn_pass1<128><<<64, blk, 0, stream>>>(q2, k2, m2, s2, N);
  attn_pass2<128><<<64, blk, 0, stream>>>(q2, k2, v2T, m2, s2, o2, N);
  // 7. edge_ctx = X2 + (o2 @ f0^T + b)  -> f32 out + f16 staging
  gemm_wmma<true,true,true,true,false><<<dim3(2, 64), blk, 0, stream>>>(
      o2, 128, W2f0, 128, r2_f0b, 256, X2, 256, edge_ctx, 256, ec16, 256, nullptr, 0, 128);
  // 8. edge_rep = edge_ctx @ f1^T + b   (M=8192, Npad=8192, Kpad=256)
  gemm_wmma<true,false,true,false,false><<<dim3(64, 64), blk, 0, stream>>>(
      ec16, 256, W2f1, 256, r2_f1b, 8192, nullptr, 0, edge_rep, 8192, nullptr, 0, nullptr, 0, 256);
}
